// GraphNetBlock_35201551958677
// MI455X (gfx1250) — compile-verified, matrix operand order checked by reading, not run on hardware
//
#include <hip/hip_runtime.h>
#include <hip/hip_bf16.h>

typedef __attribute__((ext_vector_type(16))) __bf16 v16bf;
typedef __attribute__((ext_vector_type(8)))  float  v8f;

#define WAVES 4
#define D 128
#define HB_STRIDE 136   // bf16 activation row stride (272 B, 16B-aligned rows)
#define HF_STRIDE 132   // f32 activation row stride  (528 B, 16B-aligned rows)

// ---- helpers -------------------------------------------------------------

__device__ __forceinline__ unsigned int rne_bf16(unsigned int xb) {
  return (xb + 0x7FFFu + ((xb >> 16) & 1u)) >> 16;   // round-to-nearest-even
}

// pack two f32 -> bf16x2 with round-half-up, via single v_perm_b32
__device__ __forceinline__ unsigned int pack_bf16_rh(float x, float y) {
#if __has_builtin(__builtin_amdgcn_perm)
  const unsigned int xb = __float_as_uint(x) + 0x8000u;
  const unsigned int yb = __float_as_uint(y) + 0x8000u;
  // result = { yb[31:24], yb[23:16], xb[31:24], xb[23:16] }
  return __builtin_amdgcn_perm(yb, xb, 0x07060302u);
#else
  return (rne_bf16(__float_as_uint(y)) << 16) |
         (rne_bf16(__float_as_uint(x)) & 0xFFFFu);
#endif
}

__device__ __forceinline__ void atomic_add_f32(float* p, float v) {
  __hip_atomic_fetch_add(p, v, __ATOMIC_RELAXED, __HIP_MEMORY_SCOPE_AGENT);
}

// ---- weight pre-swizzle: f32 [K][128] -> fragment-major bf16 -------------
// per-MLP layout: [L1: 12 ktiles][L2: 4][L3: 4]; ktile = 8nt * 32lane * 16e
// 16-bit A/B fragment element e of lane (half=lane>>4):
//   K(e) = (e<8) ? half*8 + e : 16 + half*8 + (e-8)   (contiguous 8-runs)
__global__ void mgn_convert_weights(const float* __restrict__ W1,
                                    const float* __restrict__ W2,
                                    const float* __restrict__ W3,
                                    unsigned short* __restrict__ outw) {
  const int perL1 = 12 * 8 * 32 * 16;    // 49152
  const int perL23 = 4 * 8 * 32 * 16;    // 16384
  const int perMlp = perL1 + 2 * perL23; // 81920
  int idx = blockIdx.x * blockDim.x + threadIdx.x;
  if (idx >= 3 * perMlp) return;
  int mlp = idx / perMlp;
  int rem = idx - mlp * perMlp;
  const float* W;
  int r2;
  if (rem < perL1)               { W = W1 + (size_t)mlp * 384 * 128; r2 = rem; }
  else if (rem < perL1 + perL23) { W = W2 + (size_t)mlp * 128 * 128; r2 = rem - perL1; }
  else                           { W = W3 + (size_t)mlp * 128 * 128; r2 = rem - perL1 - perL23; }
  int e    = r2 & 15;
  int lane = (r2 >> 4) & 31;
  int nt   = (r2 >> 9) & 7;
  int kt   = r2 >> 12;
  int half = lane >> 4;
  int k = kt * 32 + ((e < 8) ? (half * 8 + e) : (16 + half * 8 + (e - 8)));
  int n = nt * 16 + (lane & 15);
  outw[idx] = (unsigned short)rne_bf16(__float_as_uint(W[(size_t)k * 128 + n]));
}

// ---- node features f32 -> bf16 copy (streaming, 8 elems/thread) ----------
__global__ void mgn_feat_to_bf16(const float* __restrict__ in,
                                 unsigned int* __restrict__ out, int n8) {
  int i = blockIdx.x * blockDim.x + threadIdx.x;
  if (i >= n8) return;
  const float4 f0 = *(const float4*)(in + (size_t)i * 8);
  const float4 f1 = *(const float4*)(in + (size_t)i * 8 + 4);
  uint4 o;
  o.x = pack_bf16_rh(f0.x, f0.y);
  o.y = pack_bf16_rh(f0.z, f0.w);
  o.z = pack_bf16_rh(f1.x, f1.y);
  o.w = pack_bf16_rh(f1.z, f1.w);
  *(uint4*)(out + (size_t)i * 4) = o;
}

// ---- fused gather -> MLP(384,128,128) -> LayerNorm -> residual/scatter ---
// EDGE=true : x = [nodebf[sender], nodebf[recv], f2_edgefeat[row]]
// EDGE=false: x = [nodebf[row],    f1_aggmesh[row], f2_aggworld[row]]
template <bool EDGE>
__global__ __launch_bounds__(WAVES * 32)
void mgn_mlp_kernel(const unsigned short* __restrict__ nbf,
                    const float* __restrict__ f1, const float* __restrict__ f2,
                    const int* __restrict__ sidx, const int* __restrict__ ridx,
                    const unsigned short* __restrict__ wfrag,
                    const float* __restrict__ bias1, const float* __restrict__ bias2,
                    const float* __restrict__ bias3, const float* __restrict__ ln_g,
                    const float* __restrict__ ln_b, const float* __restrict__ resid,
                    float* __restrict__ out, float* __restrict__ agg, int nRows) {
  __shared__ unsigned short hb_s[WAVES][16 * HB_STRIDE];
  __shared__ float          hf_s[WAVES][16 * HF_STRIDE];

  const int lane = threadIdx.x & 31;
  const int wave = threadIdx.x >> 5;
  const int half = lane >> 4;
  const int mrow = lane & 15;

  int r0 = (blockIdx.x * WAVES + wave) * 16;
  const bool active = (r0 < nRows);
  if (!active) r0 = (nRows >= 16) ? nRows - 16 : 0;  // row counts are /16

  unsigned short* hb = hb_s[wave];
  float*          hf = hf_s[wave];

  const int row = r0 + mrow;
  const int si = EDGE ? sidx[row] : row;
  const int ri = EDGE ? ridx[row] : row;

  const unsigned short* bbase0 = nbf + (size_t)si * D;             // ktiles 0-3
  const unsigned short* bbase1 = EDGE ? nbf + (size_t)ri * D : nullptr;
  const float*          fbase1 = EDGE ? nullptr : f1 + (size_t)row * D;
  const float*          fbase2 = f2 + (size_t)row * D;             // ktiles 8-11

  v8f acc[8];
#pragma unroll
  for (int nt = 0; nt < 8; ++nt)
#pragma unroll
    for (int v = 0; v < 8; ++v) acc[nt][v] = 0.0f;

  const unsigned short* w1 = wfrag;
  // one 32-wide k-step: A-fragment `a` x 8 column tiles of weights
  auto gemm_step = [&](const v16bf& av, int ktw) {
#pragma unroll
    for (int nt = 0; nt < 8; ++nt) {
      union { v16bf v; uint4 q[2]; } b;
      const uint4* bp = (const uint4*)(w1 + ((size_t)(ktw * 8 + nt) * 32 + lane) * 16);
      b.q[0] = bp[0]; b.q[1] = bp[1];
      acc[nt] = __builtin_amdgcn_wmma_f32_16x16x32_bf16(
          false, av, false, b.v, (short)0, acc[nt], false, false);
    }
  };
  // A-fragment from row-major bf16: two contiguous b128 runs per lane
  auto load_a_bf16 = [&](const unsigned short* base, int kt) {
    union { v16bf v; uint4 q[2]; } a;
    const unsigned short* p = base + kt * 32 + half * 8;
    a.q[0] = *(const uint4*)(p);
    a.q[1] = *(const uint4*)(p + 16);
    return a.v;
  };
  // A-fragment from row-major f32: 4x b128 + perm-pack
  auto load_a_f32 = [&](const float* base, int kt) {
    const float* p = base + kt * 32 + half * 8;
    const float4 g0 = *(const float4*)(p);
    const float4 g1 = *(const float4*)(p + 4);
    const float4 g2 = *(const float4*)(p + 16);
    const float4 g3 = *(const float4*)(p + 20);
    union { v16bf v; unsigned int u[8]; } a;
    a.u[0] = pack_bf16_rh(g0.x, g0.y);
    a.u[1] = pack_bf16_rh(g0.z, g0.w);
    a.u[2] = pack_bf16_rh(g1.x, g1.y);
    a.u[3] = pack_bf16_rh(g1.z, g1.w);
    a.u[4] = pack_bf16_rh(g2.x, g2.y);
    a.u[5] = pack_bf16_rh(g2.z, g2.w);
    a.u[6] = pack_bf16_rh(g3.x, g3.y);
    a.u[7] = pack_bf16_rh(g3.z, g3.w);
    return a.v;
  };

  // ---------- layer 1: K=384 ----------
  for (int kt = 0; kt < 4; ++kt)        // region 0: node[sender] (bf16)
    gemm_step(load_a_bf16(bbase0, kt), kt);
  for (int kt = 0; kt < 4; ++kt) {      // region 1: node[recv] (bf16) or agg (f32)
    if (EDGE) gemm_step(load_a_bf16(bbase1, kt), 4 + kt);
    else      gemm_step(load_a_f32(fbase1, kt), 4 + kt);
  }
  for (int kt = 0; kt < 4; ++kt)        // region 2: edge feats / agg_world (f32)
    gemm_step(load_a_f32(fbase2, kt), 8 + kt);

  // bias + ReLU -> LDS bf16 (C/D layout: n = 16*nt + mrow, m = 8*half + v)
#pragma unroll
  for (int nt = 0; nt < 8; ++nt) {
    const int col = nt * 16 + mrow;
    const float bv = bias1[col];
#pragma unroll
    for (int v = 0; v < 8; v += 2) {
      const float x0 = fmaxf(acc[nt][v] + bv, 0.0f);
      const float x1 = fmaxf(acc[nt][v + 1] + bv, 0.0f);
      const unsigned int p = pack_bf16_rh(x0, x1);
      hb[(half * 8 + v) * HB_STRIDE + col]     = (unsigned short)(p & 0xFFFFu);
      hb[(half * 8 + v + 1) * HB_STRIDE + col] = (unsigned short)(p >> 16);
    }
  }
  __syncthreads();

  // ---------- layers 2 & 3: K=128, A-fragments from LDS (2x b128 each) ----
  const unsigned short* wl[2] = { wfrag + 12 * 8 * 32 * 16,
                                  wfrag + 12 * 8 * 32 * 16 + 4 * 8 * 32 * 16 };
  const float* bl[2] = { bias2, bias3 };
#pragma unroll 1
  for (int layer = 0; layer < 2; ++layer) {
#pragma unroll
    for (int nt = 0; nt < 8; ++nt)
#pragma unroll
      for (int v = 0; v < 8; ++v) acc[nt][v] = 0.0f;
    const unsigned short* w = wl[layer];
    for (int kt = 0; kt < 4; ++kt) {
      union { v16bf v; uint4 q[2]; } a;
      const unsigned short* hrow0 = hb + mrow * HB_STRIDE + kt * 32 + half * 8;
      a.q[0] = *(const uint4*)(hrow0);
      a.q[1] = *(const uint4*)(hrow0 + 16);
#pragma unroll
      for (int nt = 0; nt < 8; ++nt) {
        union { v16bf v; uint4 q[2]; } b;
        const uint4* bp = (const uint4*)(w + ((size_t)(kt * 8 + nt) * 32 + lane) * 16);
        b.q[0] = bp[0]; b.q[1] = bp[1];
        acc[nt] = __builtin_amdgcn_wmma_f32_16x16x32_bf16(
            false, a.v, false, b.v, (short)0, acc[nt], false, false);
      }
    }
    __syncthreads();  // all LDS reads done before overwrite
    if (layer == 0) {
#pragma unroll
      for (int nt = 0; nt < 8; ++nt) {
        const int col = nt * 16 + mrow;
        const float bv = bl[0][col];
#pragma unroll
        for (int v = 0; v < 8; v += 2) {
          const float x0 = fmaxf(acc[nt][v] + bv, 0.0f);
          const float x1 = fmaxf(acc[nt][v + 1] + bv, 0.0f);
          const unsigned int p = pack_bf16_rh(x0, x1);
          hb[(half * 8 + v) * HB_STRIDE + col]     = (unsigned short)(p & 0xFFFFu);
          hb[(half * 8 + v + 1) * HB_STRIDE + col] = (unsigned short)(p >> 16);
        }
      }
    } else {  // layer 3: linear out -> f32 LDS for LayerNorm
#pragma unroll
      for (int nt = 0; nt < 8; ++nt) {
        const int col = nt * 16 + mrow;
        const float bv = bl[1][col];
#pragma unroll
        for (int v = 0; v < 8; ++v)
          hf[(half * 8 + v) * HF_STRIDE + col] = acc[nt][v] + bv;
      }
    }
    __syncthreads();
  }

  // ---------- LayerNorm + residual + scatter-add ----------
  // lanes l and l^16 co-own row (l&15); each handles 64 columns
  const float* hrow = hf + mrow * HF_STRIDE + half * 64;
  float s = 0.0f, q = 0.0f;
#pragma unroll
  for (int c = 0; c < 64; c += 4) {
    const float4 x = *(const float4*)(hrow + c);
    s += x.x + x.y + x.z + x.w;
    q += x.x * x.x + x.y * x.y + x.z * x.z + x.w * x.w;
  }
  s += __shfl_xor(s, 16, 32);
  q += __shfl_xor(q, 16, 32);
  const float mean = s * (1.0f / 128.0f);
  const float var  = q * (1.0f / 128.0f) - mean * mean;
  const float inv  = rsqrtf(var + 1e-5f);

  if (active) {
    const float* res = resid + (size_t)row * D + half * 64;
    float* o  = out + (size_t)row * D + half * 64;
    float* ag = (EDGE && agg) ? agg + (size_t)ri * D + half * 64 : nullptr;
    const float* lg = ln_g + half * 64;
    const float* lb = ln_b + half * 64;
#pragma unroll 2
    for (int c = 0; c < 64; c += 4) {
      const float4 x = *(const float4*)(hrow + c);
      const float4 g = *(const float4*)(lg + c);
      const float4 bb = *(const float4*)(lb + c);
      const float4 r = *(const float4*)(res + c);
      float4 y;
      y.x = (x.x - mean) * inv * g.x + bb.x;
      y.y = (x.y - mean) * inv * g.y + bb.y;
      y.z = (x.z - mean) * inv * g.z + bb.z;
      y.w = (x.w - mean) * inv * g.w + bb.w;
      float4 ov;
      ov.x = r.x + y.x; ov.y = r.y + y.y; ov.z = r.z + y.z; ov.w = r.w + y.w;
      *(float4*)(o + c) = ov;
      if (ag) {
        atomic_add_f32(ag + c + 0, y.x);
        atomic_add_f32(ag + c + 1, y.y);
        atomic_add_f32(ag + c + 2, y.z);
        atomic_add_f32(ag + c + 3, y.w);
      }
    }
  }
}

// ---- host launcher -------------------------------------------------------

extern "C" void kernel_launch(void* const* d_in, const int* in_sizes, int n_in,
                              void* d_out, int out_size, void* d_ws, size_t ws_size,
                              hipStream_t stream) {
  const float* node    = (const float*)d_in[0];
  const float* mesh_e  = (const float*)d_in[1];
  const float* world_e = (const float*)d_in[2];
  const int*   msend   = (const int*)d_in[3];
  const int*   mrecv   = (const int*)d_in[4];
  const int*   wsend   = (const int*)d_in[5];
  const int*   wrecv   = (const int*)d_in[6];
  const float* W1 = (const float*)d_in[7];
  const float* b1 = (const float*)d_in[8];
  const float* W2 = (const float*)d_in[9];
  const float* b2 = (const float*)d_in[10];
  const float* W3 = (const float*)d_in[11];
  const float* b3 = (const float*)d_in[12];
  const float* lng = (const float*)d_in[13];
  const float* lnb = (const float*)d_in[14];

  const int N  = in_sizes[0] / D;
  const int Em = in_sizes[3];
  const int Ew = in_sizes[5];

  float* out_node  = (float*)d_out;
  float* out_mesh  = out_node + (size_t)N * D;
  float* out_world = out_mesh + (size_t)Em * D;

  // workspace: [swizzled bf16 weights][bf16 node copy][agg_mesh][agg_world]
  const int perMlp = 81920;
  unsigned short* wbf = (unsigned short*)d_ws;
  size_t off = (((size_t)3 * perMlp * 2) + 255) & ~(size_t)255;
  unsigned short* nbf = (unsigned short*)((char*)d_ws + off);
  off += ((size_t)N * D * 2 + 255) & ~(size_t)255;
  float* agg_mesh  = (float*)((char*)d_ws + off);
  float* agg_world = agg_mesh + (size_t)N * D;

  hipMemsetAsync(agg_mesh, 0, (size_t)N * D * 2 * sizeof(float), stream);

  mgn_convert_weights<<<(3 * perMlp + 255) / 256, 256, 0, stream>>>(W1, W2, W3, wbf);

  const int n8 = N * D / 8;
  mgn_feat_to_bf16<<<(n8 + 255) / 256, 256, 0, stream>>>(node, (unsigned int*)nbf, n8);

  const dim3 blk(WAVES * 32);
  const int mesh_blocks  = (Em / 16 + WAVES - 1) / WAVES;
  const int world_blocks = (Ew / 16 + WAVES - 1) / WAVES;
  const int node_blocks  = (N  / 16 + WAVES - 1) / WAVES;

  // edge MLP 0: mesh edges
  mgn_mlp_kernel<true><<<mesh_blocks, blk, 0, stream>>>(
      nbf, nullptr, mesh_e, msend, mrecv, wbf + 0 * (size_t)perMlp,
      b1 + 0, b2 + 0, b3 + 0, lng + 0, lnb + 0,
      mesh_e, out_mesh, agg_mesh, Em);
  // edge MLP 1: world edges
  mgn_mlp_kernel<true><<<world_blocks, blk, 0, stream>>>(
      nbf, nullptr, world_e, wsend, wrecv, wbf + 1 * (size_t)perMlp,
      b1 + 128, b2 + 128, b3 + 128, lng + 128, lnb + 128,
      world_e, out_world, agg_world, Ew);
  // node MLP: [node, agg_mesh, agg_world]
  mgn_mlp_kernel<false><<<node_blocks, blk, 0, stream>>>(
      nbf, agg_mesh, agg_world, nullptr, nullptr, wbf + 2 * (size_t)perMlp,
      b1 + 256, b2 + 256, b3 + 256, lng + 256, lnb + 256,
      node, out_node, nullptr, N);
}
